// MoELayer_5669356830855
// MI455X (gfx1250) — matrix operand order repaired and checked by reading, not compiled, not run on hardware
//
#include <hip/hip_runtime.h>

// ---------------- problem constants ----------------
#define DN    512
#define HN    1024
#define EN    8
#define KTOP  2
#define NTOK  65536
#define PROJN 256
#define CAP   16384                 // N*K/E
#define NSLOT (NTOK*KTOP)           // 131072
#define CHUNK 1024
#define NCHUNK (NSLOT/CHUNK)        // 128
#define LOG100 4.6051702f

#define BTP 40                      // 32 + 8 pad (bf16 elems) -> 80B rows, conflict-free
#define BTELEMS (128*BTP)           // elements per BT buffer (5120)
#define BTBYTES (BTELEMS*2)         // 10240

// Async LDS-fill path: global_load_async_to_lds_b128 + s_wait_asynccnt (CDNA5).
#define USE_ASYNC 1

// ---------------- WMMA types ----------------
typedef __attribute__((ext_vector_type(16))) __bf16 v16bf;
typedef __attribute__((ext_vector_type(8)))  float  v8f;

union FragBF { v16bf v; uint4 q[2]; };
union AccF   { v8f v; float f[8]; };

__device__ inline unsigned short f2bf(float f) {
  union { float f; unsigned u; } v; v.f = f;
  unsigned u = v.u;
  return (unsigned short)((u + 0x7FFFu + ((u >> 16) & 1u)) >> 16);
}

__device__ inline v8f acc_zero() {
  v8f z;
#pragma unroll
  for (int i = 0; i < 8; ++i) z[i] = 0.0f;
  return z;
}

__device__ inline v8f wmma_bf16(v16bf a, v16bf b, v8f c) {
  return __builtin_amdgcn_wmma_f32_16x16x32_bf16(false, a, false, b, (short)0, c, false, false);
}

// A-frag (16x32 bf16) per ISA 7.12.2: lane&15 = M row; lanes<16 hold K {0-7,16-23},
// lanes>=16 hold K {8-15,24-31}. Two 16B runs per lane.
__device__ inline v16bf load_a_lds(const unsigned short* A, int pitch, int m, int k, int lane) {
  const unsigned short* p = A + (size_t)(m + (lane & 15)) * pitch + k + ((lane >> 4) << 3);
  FragBF u;
  u.q[0] = *(const uint4*)(p);
  u.q[1] = *(const uint4*)(p + 16);
  return u.v;
}

// B-frag (32x16 bf16) from k-major LDS tile BT[n][k] (pitch BTP), buffer chosen by
// element offset eoff (single base pointer keeps addrspace(3) inference intact):
// lane&15 = N col, lanes<16 K {0-7,16-23}, lanes>=16 K {8-15,24-31} -> two 16B runs.
__device__ inline v16bf load_bT_frag(const unsigned short* BT, unsigned eoff, int n, int lane) {
  const unsigned short* p = BT + eoff + (size_t)(n + (lane & 15)) * BTP + ((lane >> 4) << 3);
  FragBF u;
  u.q[0] = *(const uint4*)(p);
  u.q[1] = *(const uint4*)(p + 16);
  return u.v;
}

// ---------------- B-chunk staging: async global->LDS copy (ASYNCcnt) ----------------
#if USE_ASYNC
__device__ inline void async_cp16(unsigned lds_off, const void* gaddr) {
  asm volatile("global_load_async_to_lds_b128 %0, %1, off"
               :: "v"(lds_off), "v"(gaddr)
               : "memory");
}
__device__ inline void stage_fence() {
  asm volatile("s_wait_asynccnt 0" ::: "memory");
}
#else
__device__ inline void stage_fence() {}
#endif

// copy [128 rows x 32 elem] bf16 tile (row stride = rowlen elems) into BT buffer
// (element offset eoff, LDS byte offset lds_off)
__device__ inline void stage_bt(unsigned short* BT, unsigned eoff, unsigned lds_off,
                                const unsigned short* src, int rowlen, int tid) {
  int n = tid >> 1, half = tid & 1;
  const uint4* s = (const uint4*)(src + (size_t)n * rowlen) + half * 2;
#if USE_ASYNC
  (void)BT; (void)eoff;
  unsigned d = lds_off + (unsigned)n * (BTP * 2) + (unsigned)half * 32;
  async_cp16(d, s);
  async_cp16(d + 16, s + 1);
#else
  uint4* dptr = (uint4*)(BT + eoff + (size_t)n * BTP) + half * 2;
  dptr[0] = s[0];
  dptr[1] = s[1];
#endif
}

// ---------------- tiny utility kernels ----------------
__global__ void moe5_zero_f32(float* p, int n) {
  int i = blockIdx.x * blockDim.x + threadIdx.x;
  if (i < n) p[i] = 0.0f;
}
__global__ void moe5_fill_i32(int* p, int n, int v) {
  int i = blockIdx.x * blockDim.x + threadIdx.x;
  if (i < n) p[i] = v;
}
__global__ void moe5_cvt_bf16(const float* __restrict__ s, unsigned short* __restrict__ d, int n) {
  int i = blockIdx.x * blockDim.x + threadIdx.x;
  if (i < n) d[i] = f2bf(s[i]);
}

// tiled transpose + f32->bf16: out[c][r] = bf16(in[r][c]); batch via blockIdx.z
__global__ void moe5_cvtT(const float* __restrict__ in, unsigned short* __restrict__ out,
                          int R, int Cc) {
  __shared__ float tile[32][33];
  const float* inb = in + (size_t)blockIdx.z * R * Cc;
  unsigned short* outb = out + (size_t)blockIdx.z * R * Cc;
  int c0 = blockIdx.x * 32, r0 = blockIdx.y * 32;
  int tx = threadIdx.x & 31, ty = threadIdx.x >> 5;
  for (int i = ty; i < 32; i += 8)
    tile[i][tx] = inb[(size_t)(r0 + i) * Cc + c0 + tx];
  __syncthreads();
  for (int i = ty; i < 32; i += 8)
    outb[(size_t)(c0 + i) * R + r0 + tx] = f2bf(tile[tx][i]);
}

// normalize each of the EN columns of sim [PROJN][EN]
__global__ void moe5_simhat(const float* __restrict__ sim, float* __restrict__ simhat) {
  __shared__ float ssq[EN];
  int e = threadIdx.x >> 5, lane = threadIdx.x & 31;
  float s = 0.f;
  for (int i = lane; i < PROJN; i += 32) { float v = sim[i * EN + e]; s += v * v; }
  for (int off = 16; off; off >>= 1) s += __shfl_down(s, off, 32);
  if (lane == 0) ssq[e] = s;
  __syncthreads();
  float inv = 1.0f / (sqrtf(ssq[e]) + 1e-12f);
  for (int i = lane; i < PROJN; i += 32) simhat[i * EN + e] = sim[i * EN + e] * inv;
}

// ---------------- fused gate kernel ----------------
#define GP_PITCH 260
#define G_PL_B   (128*GP_PITCH*4)         // 133120
#define G_BT0    G_PL_B
#define G_SIM    (G_PL_B + 2*BTBYTES)
#define G_BP     (G_SIM + PROJN*EN*4)
#define GATE_SHM (G_BP + PROJN*4)

__global__ void __launch_bounds__(256, 1)
moe5_gate(const unsigned short* __restrict__ xbf,
          const unsigned short* __restrict__ WpT,  // [PROJN][DN] bf16
          const float* __restrict__ bp,
          const float* __restrict__ simhat,
          const float* __restrict__ temp,
          const int* __restrict__ label,
          int mode,
          int* __restrict__ tokE, float* __restrict__ tokG) {
  extern __shared__ char smem[];
  float* Pl = (float*)smem;
  unsigned short* BT = (unsigned short*)(smem + G_BT0);   // 2 buffers of BTELEMS
  float* sh_sim = (float*)(smem + G_SIM);
  float* sh_bp  = (float*)(smem + G_BP);

  int tid = threadIdx.x;
  int t0 = blockIdx.x * 128;
  for (int i = tid; i < PROJN; i += 256) sh_bp[i] = bp[i];
  for (int i = tid; i < PROJN * EN; i += 256) sh_sim[i] = simhat[i];

  int lane = tid & 31, w = tid >> 5;
  int wm = (w & 3) * 32, wn = (w >> 2) * 64;
  int kh = (lane >> 4) << 3;
  const unsigned short* a0p = xbf + (size_t)(t0 + wm + (lane & 15)) * DN;
  const unsigned short* a1p = a0p + (size_t)16 * DN;

  for (int nc = 0; nc < PROJN / 128; ++nc) {
    int n0 = nc * 128;
    const unsigned short* wsrc = WpT + (size_t)n0 * DN;
    v8f acc[2][4];
#pragma unroll
    for (int i = 0; i < 2; ++i)
#pragma unroll
      for (int j = 0; j < 4; ++j) acc[i][j] = acc_zero();

    stage_bt(BT, 0u, G_BT0, wsrc, DN, tid);
#pragma unroll 2
    for (int ki = 0; ki < DN / 32; ++ki) {
      int ks = ki * 32, cur = ki & 1;
      unsigned ce = (unsigned)cur * BTELEMS;
      stage_fence();
      __syncthreads();
      if (ki + 1 < DN / 32) {
        unsigned cn = (unsigned)(cur ^ 1) * BTELEMS;
        stage_bt(BT, cn, G_BT0 + 2u * cn, wsrc + ks + 32, DN, tid);
      }
      FragBF a0u, a1u;
      a0u.q[0] = *(const uint4*)(a0p + ks + kh);
      a0u.q[1] = *(const uint4*)(a0p + ks + kh + 16);
      a1u.q[0] = *(const uint4*)(a1p + ks + kh);
      a1u.q[1] = *(const uint4*)(a1p + ks + kh + 16);
#pragma unroll
      for (int j = 0; j < 4; ++j) {
        v16bf b = load_bT_frag(BT, ce, wn + j * 16, lane);
        acc[0][j] = wmma_bf16(a0u.v, b, acc[0][j]);
        acc[1][j] = wmma_bf16(a1u.v, b, acc[1][j]);
      }
    }
#pragma unroll
    for (int i = 0; i < 2; ++i)
#pragma unroll
      for (int j = 0; j < 4; ++j) {
        AccF c; c.v = acc[i][j];
        int colg = n0 + wn + j * 16 + (lane & 15);
        float bb = sh_bp[colg];
#pragma unroll
        for (int r = 0; r < 8; ++r) {
          int row = wm + i * 16 + r + ((lane >> 4) << 3);
          Pl[(size_t)row * GP_PITCH + colg] = c.f[r] + bb;
        }
      }
  }
  __syncthreads();

  if (tid < 128) {
    int token = t0 + tid;
    const float* Prow = Pl + (size_t)tid * GP_PITCH;
    float ss = 0.f;
    for (int i = 0; i < PROJN; ++i) { float v = Prow[i]; ss += v * v; }
    float inv = 1.0f / (sqrtf(ss) + 1e-12f);
    float scale = expf(fminf(temp[0], LOG100));
    float le[EN];
#pragma unroll
    for (int e = 0; e < EN; ++e) le[e] = 0.f;
    for (int i = 0; i < PROJN; ++i) {
      float p = Prow[i];
#pragma unroll
      for (int e = 0; e < EN; ++e) le[e] += p * sh_sim[i * EN + e];
    }
    float mx = -1e30f;
#pragma unroll
    for (int e = 0; e < EN; ++e) { le[e] *= inv * scale; mx = fmaxf(mx, le[e]); }
    float se = 0.f;
#pragma unroll
    for (int e = 0; e < EN; ++e) { le[e] = expf(le[e] - mx); se += le[e]; }
    int e0 = 0; float g0 = le[0];
#pragma unroll
    for (int e = 1; e < EN; ++e) if (le[e] > g0) { g0 = le[e]; e0 = e; }
    int e1 = -1; float g1 = -1.f;
#pragma unroll
    for (int e = 0; e < EN; ++e) if (e != e0 && le[e] > g1) { g1 = le[e]; e1 = e; }
    g0 /= se; g1 /= se;
    float gs = g0 + g1 + 1e-9f;
    g0 /= gs; g1 /= gs;
    int lbl = label[token];
    int valid = (mode == 0) || (mode == 1 && lbl == 0) || (mode == 2 && lbl == 1);
    if (valid) {
      tokE[2 * token] = e0;  tokG[2 * token] = g0;
      tokE[2 * token + 1] = e1; tokG[2 * token + 1] = g1;
    } else {
      tokE[2 * token] = -1; tokE[2 * token + 1] = -1;
      tokG[2 * token] = 0.f; tokG[2 * token + 1] = 0.f;
    }
  }
}

// ---------------- routing (deterministic token-order cumsum) ----------------
__global__ void moe5_hist(const int* __restrict__ tokE, int* __restrict__ hist) {
  __shared__ int cnt[EN];
  if (threadIdx.x < EN) cnt[threadIdx.x] = 0;
  __syncthreads();
  int base = blockIdx.x * CHUNK;
  for (int i = threadIdx.x; i < CHUNK; i += blockDim.x) {
    int e = tokE[base + i];
    if (e >= 0) atomicAdd(&cnt[e], 1);
  }
  __syncthreads();
  if (threadIdx.x < EN) hist[blockIdx.x * EN + threadIdx.x] = cnt[threadIdx.x];
}

__global__ void moe5_scan(const int* __restrict__ hist, int* __restrict__ coff) {
  int e = threadIdx.x;
  if (e < EN) {
    int run = 0;
    for (int c = 0; c < NCHUNK; ++c) { coff[c * EN + e] = run; run += hist[c * EN + e]; }
  }
}

__global__ void moe5_assign(const int* __restrict__ tokE, const float* __restrict__ tokG,
                            const int* __restrict__ coff,
                            int* __restrict__ idxt, float* __restrict__ gatet) {
  int chunk = blockIdx.x;
  int lane = threadIdx.x;
  int cnt[EN];
#pragma unroll
  for (int e = 0; e < EN; ++e) cnt[e] = coff[chunk * EN + e];
  unsigned lt = (1u << lane) - 1u;
  int base = chunk * CHUNK;
  for (int it = 0; it < CHUNK / 32; ++it) {
    int s = base + it * 32 + lane;
    int e = tokE[s];
    float g = tokG[s];
    int pos = -1;
#pragma unroll
    for (int ee = 0; ee < EN; ++ee) {
      unsigned m = (unsigned)__ballot(e == ee);
      if (e == ee) pos = cnt[ee] + __popc(m & lt);
      cnt[ee] += __popc(m);
    }
    if (e >= 0 && pos < CAP) {
      int tok = s >> 1;
      idxt[e * CAP + pos] = tok;
      gatet[e * CAP + pos] = g;
    }
  }
}

// ---------------- fused expert FFN ----------------
#define FH_PITCH 1032
#define F_HS_B   (128*FH_PITCH*2)         // 264192
#define F_BT0    F_HS_B
#define F_TOK    (F_BT0 + 2*BTBYTES)
#define F_GAT    (F_TOK + 128*4)
#define FFN_SHM  (F_GAT + 128*4)

__global__ void __launch_bounds__(256, 1)
moe5_ffn(const unsigned short* __restrict__ xbf,
         const unsigned short* __restrict__ W1T,  // [E][HN][DN] bf16
         const float* __restrict__ b1,
         const unsigned short* __restrict__ W2T,  // [E][DN][HN] bf16
         const float* __restrict__ b2,
         const int* __restrict__ idxt,
         const float* __restrict__ gatet,
         float* __restrict__ out,
         const float* __restrict__ scale_p, int use_scale) {
  extern __shared__ char smem[];
  unsigned short* Hs = (unsigned short*)smem;
  unsigned short* BT = (unsigned short*)(smem + F_BT0);   // 2 buffers of BTELEMS
  int* tokid = (int*)(smem + F_TOK);
  float* gatev = (float*)(smem + F_GAT);

  int tid = threadIdx.x;
  int e = blockIdx.x >> 7;
  int row0 = (blockIdx.x & 127) * 128;
  const unsigned short* W1e = W1T + (size_t)e * DN * HN;
  const unsigned short* W2e = W2T + (size_t)e * HN * DN;
  const float* b1e = b1 + (size_t)e * HN;
  const float* b2e = b2 + (size_t)e * DN;
  float mscale = use_scale ? scale_p[0] : 1.0f;

  if (tid < 128) {
    tokid[tid] = idxt[(size_t)e * CAP + row0 + tid];
    gatev[tid] = gatet[(size_t)e * CAP + row0 + tid];
  }
  __syncthreads();

  int lane = tid & 31, w = tid >> 5;
  int wm = (w & 3) * 32, wn = (w >> 2) * 64;
  int kh = (lane >> 4) << 3;
  int tr0 = tokid[wm + (lane & 15)];      if (tr0 < 0) tr0 = 0;
  int tr1 = tokid[wm + 16 + (lane & 15)]; if (tr1 < 0) tr1 = 0;
  const unsigned short* a0p = xbf + (size_t)tr0 * DN;
  const unsigned short* a1p = xbf + (size_t)tr1 * DN;

  // ---- GEMM1: h = relu(gather(x) @ W1 + b1) -> bf16 tile in LDS ----
  for (int nc = 0; nc < HN / 128; ++nc) {
    int n0 = nc * 128;
    const unsigned short* wsrc = W1e + (size_t)n0 * DN;
    v8f acc[2][4];
#pragma unroll
    for (int i = 0; i < 2; ++i)
#pragma unroll
      for (int j = 0; j < 4; ++j) acc[i][j] = acc_zero();

    stage_bt(BT, 0u, F_BT0, wsrc, DN, tid);
#pragma unroll 2
    for (int ki = 0; ki < DN / 32; ++ki) {
      int ks = ki * 32, cur = ki & 1;
      unsigned ce = (unsigned)cur * BTELEMS;
      stage_fence();
      __syncthreads();
      if (ki + 1 < DN / 32) {
        unsigned cn = (unsigned)(cur ^ 1) * BTELEMS;
        stage_bt(BT, cn, F_BT0 + 2u * cn, wsrc + ks + 32, DN, tid);
      }
      FragBF a0u, a1u;
      a0u.q[0] = *(const uint4*)(a0p + ks + kh);
      a0u.q[1] = *(const uint4*)(a0p + ks + kh + 16);
      a1u.q[0] = *(const uint4*)(a1p + ks + kh);
      a1u.q[1] = *(const uint4*)(a1p + ks + kh + 16);
#pragma unroll
      for (int j = 0; j < 4; ++j) {
        v16bf b = load_bT_frag(BT, ce, wn + j * 16, lane);
        acc[0][j] = wmma_bf16(a0u.v, b, acc[0][j]);
        acc[1][j] = wmma_bf16(a1u.v, b, acc[1][j]);
      }
    }
#pragma unroll
    for (int i = 0; i < 2; ++i)
#pragma unroll
      for (int j = 0; j < 4; ++j) {
        AccF c; c.v = acc[i][j];
        int colg = n0 + wn + j * 16 + (lane & 15);
        float bb = b1e[colg];
#pragma unroll
        for (int r = 0; r < 8; ++r) {
          int row = wm + i * 16 + r + ((lane >> 4) << 3);
          float v = c.f[r] + bb;
          Hs[(size_t)row * FH_PITCH + colg] = f2bf(v > 0.f ? v : 0.f);
        }
      }
  }
  __syncthreads();

  // ---- GEMM2: y = h @ W2 + b2, gated atomic scatter ----
  for (int nc = 0; nc < DN / 128; ++nc) {
    int n0 = nc * 128;
    const unsigned short* wsrc = W2e + (size_t)n0 * HN;
    v8f acc[2][4];
#pragma unroll
    for (int i = 0; i < 2; ++i)
#pragma unroll
      for (int j = 0; j < 4; ++j) acc[i][j] = acc_zero();

    stage_bt(BT, 0u, F_BT0, wsrc, HN, tid);
#pragma unroll 2
    for (int ki = 0; ki < HN / 32; ++ki) {
      int ks = ki * 32, cur = ki & 1;
      unsigned ce = (unsigned)cur * BTELEMS;
      stage_fence();
      __syncthreads();
      if (ki + 1 < HN / 32) {
        unsigned cn = (unsigned)(cur ^ 1) * BTELEMS;
        stage_bt(BT, cn, F_BT0 + 2u * cn, wsrc + ks + 32, HN, tid);
      }
      v16bf a0 = load_a_lds(Hs, FH_PITCH, wm, ks, lane);
      v16bf a1 = load_a_lds(Hs, FH_PITCH, wm + 16, ks, lane);
#pragma unroll
      for (int j = 0; j < 4; ++j) {
        v16bf b = load_bT_frag(BT, ce, wn + j * 16, lane);
        acc[0][j] = wmma_bf16(a0, b, acc[0][j]);
        acc[1][j] = wmma_bf16(a1, b, acc[1][j]);
      }
    }
#pragma unroll
    for (int i = 0; i < 2; ++i)
#pragma unroll
      for (int j = 0; j < 4; ++j) {
        AccF c; c.v = acc[i][j];
        int colg = n0 + wn + j * 16 + (lane & 15);
        float bb = b2e[colg];
#pragma unroll
        for (int r = 0; r < 8; ++r) {
          int row = wm + i * 16 + r + ((lane >> 4) << 3);
          int t = tokid[row];
          if (t >= 0) {
            float gv = gatev[row] * mscale;
            atomicAdd(out + (size_t)t * DN + colg, gv * (c.f[r] + bb));
          }
        }
      }
  }
}

// ---------------- host launcher ----------------
extern "C" void kernel_launch(void* const* d_in, const int* in_sizes, int n_in,
                              void* d_out, int out_size, void* d_ws, size_t ws_size,
                              hipStream_t stream) {
  (void)in_sizes; (void)n_in; (void)out_size; (void)ws_size;
  const float* x = (const float*)d_in[0];
  const int* label = (const int*)d_in[1];
  const float* dscale = (const float*)d_in[2];
  struct P { const float *Wp, *bp, *sim, *temp, *W1, *b1, *W2, *b2; };
  P prm[3];
  for (int m = 0; m < 3; ++m) {
    int b = 3 + m * 8;
    prm[m].Wp = (const float*)d_in[b + 0];
    prm[m].bp = (const float*)d_in[b + 1];
    prm[m].sim = (const float*)d_in[b + 2];
    prm[m].temp = (const float*)d_in[b + 3];
    prm[m].W1 = (const float*)d_in[b + 4];
    prm[m].b1 = (const float*)d_in[b + 5];
    prm[m].W2 = (const float*)d_in[b + 6];
    prm[m].b2 = (const float*)d_in[b + 7];
  }
  float* out = (float*)d_out;

  char* wp = (char*)d_ws;
  auto take = [&](size_t bytes) -> char* {
    char* p = wp; wp += (bytes + 255) & ~(size_t)255; return p;
  };
  unsigned short* xbf = (unsigned short*)take((size_t)NTOK * DN * 2);
  unsigned short* WpT = (unsigned short*)take((size_t)DN * PROJN * 2);
  unsigned short* W1T = (unsigned short*)take((size_t)EN * DN * HN * 2);
  unsigned short* W2T = (unsigned short*)take((size_t)EN * HN * DN * 2);
  float* simhat = (float*)take(PROJN * EN * 4);
  int*   tokE   = (int*)take((size_t)NSLOT * 4);
  float* tokG   = (float*)take((size_t)NSLOT * 4);
  int*   hist   = (int*)take(NCHUNK * EN * 4);
  int*   coff   = (int*)take(NCHUNK * EN * 4);
  int*   idxt   = (int*)take((size_t)EN * CAP * 4);
  float* gatet  = (float*)take((size_t)EN * CAP * 4);

  const int TPB = 256;
  moe5_zero_f32<<<(NTOK * DN + TPB - 1) / TPB, TPB, 0, stream>>>(out, NTOK * DN);
  moe5_cvt_bf16<<<(NTOK * DN + TPB - 1) / TPB, TPB, 0, stream>>>(x, xbf, NTOK * DN);

  for (int m = 0; m < 3; ++m) {
    moe5_cvtT<<<dim3(PROJN / 32, DN / 32, 1), 256, 0, stream>>>(prm[m].Wp, WpT, DN, PROJN);
    moe5_cvtT<<<dim3(HN / 32, DN / 32, EN), 256, 0, stream>>>(prm[m].W1, W1T, DN, HN);
    moe5_cvtT<<<dim3(DN / 32, HN / 32, EN), 256, 0, stream>>>(prm[m].W2, W2T, HN, DN);
    moe5_simhat<<<1, 256, 0, stream>>>(prm[m].sim, simhat);

    moe5_gate<<<NTOK / 128, 256, GATE_SHM, stream>>>(xbf, WpT, prm[m].bp, simhat,
                                                     prm[m].temp, label, m, tokE, tokG);
    moe5_hist<<<NCHUNK, 256, 0, stream>>>(tokE, hist);
    moe5_scan<<<1, 32, 0, stream>>>(hist, coff);
    moe5_fill_i32<<<(EN * CAP + TPB - 1) / TPB, TPB, 0, stream>>>(idxt, EN * CAP, -1);
    moe5_assign<<<NCHUNK, 32, 0, stream>>>(tokE, tokG, coff, idxt, gatet);

    moe5_ffn<<<EN * (CAP / 128), 256, FFN_SHM, stream>>>(xbf, W1T, prm[m].b1, W2T, prm[m].b2,
                                                         idxt, gatet, out, dscale,
                                                         m == 0 ? 0 : 1);
  }
}
